// CQCNNPieceEstimator_61392262529335
// MI455X (gfx1250) — compile-verified, matrix operand order checked by reading, not run on hardware
//
#include <hip/hip_runtime.h>
#include <hip/hip_bf16.h>
#include <math.h>

// ---------------------------------------------------------------------------
// MI455X (gfx1250) implementation.
//  * Heads (feats@W) run on v_wmma_f32_16x16x32_f16, f32 accumulate.
//  * Weight matrices are prepacked once into the exact 16-bit B-fragment
//    lane/VGPR layout (ISA 7.12.2) so the GEMM loop does plain dword loads.
//  * Convs / quantum sim run on VALU with f16 LDS staging (compute-light,
//    L2-resident weights).
// ---------------------------------------------------------------------------

typedef __attribute__((ext_vector_type(16))) _Float16 v16h;
typedef __attribute__((ext_vector_type(8)))  float    v8f;

union HFrag { v16h v; _Float16 h[16]; unsigned u[8]; };

// k-offset inside a 16x32 f16 fragment for VGPR index fv and lane-half lh.
__device__ __forceinline__ int kb_of(int fv, int lh) {
    return ((fv >= 4) ? 16 : 0) + lh * 8 + (fv & 3) * 2;
}

// Build A fragment (16 x 32, f16) from an LDS row (row = &feat[M*stride]).
__device__ __forceinline__ v16h load_a_frag(const _Float16* row, int k0, int lh) {
    HFrag f;
#pragma unroll
    for (int fv = 0; fv < 8; ++fv) {
        int kb = kb_of(fv, lh);
        f.h[2 * fv]     = row[k0 + kb];
        f.h[2 * fv + 1] = row[k0 + kb + 1];
    }
    return f.v;
}

// Load prepacked B fragment: layout [tile][lane][fv] of packed f16 pairs.
__device__ __forceinline__ v16h load_b_frag(const unsigned* __restrict__ pk,
                                            int tile, int lane) {
    HFrag f;
    const unsigned* p = pk + (size_t)tile * 256 + lane * 8;
#pragma unroll
    for (int fv = 0; fv < 8; ++fv) f.u[fv] = p[fv];
    return f.v;
}

// ---------------------------------------------------------------------------
// Prepack: f32 KxN row-major weight -> f16 B fragments (pad K with zeros).
// grid = KT*NT tiles, block = 256 (32 lanes x 8 fragment VGPRs).
// ---------------------------------------------------------------------------
__global__ void pack_wmma_b(const float* __restrict__ src, int Kreal, int N,
                            int NT, unsigned* __restrict__ dst) {
    int tile = blockIdx.x;
    int kt = tile / NT, nt = tile % NT;
    int lane = threadIdx.x >> 3;
    int fv   = threadIdx.x & 7;
    int lh   = lane >> 4;
    int kb   = kb_of(fv, lh);
    int k = kt * 32 + kb;
    int n = nt * 16 + (lane & 15);
    float lo = (k     < Kreal) ? src[(size_t)k * N + n]       : 0.f;
    float hi = (k + 1 < Kreal) ? src[(size_t)(k + 1) * N + n] : 0.f;
    union { _Float16 h[2]; unsigned u; } p;
    p.h[0] = (_Float16)lo; p.h[1] = (_Float16)hi;
    dst[(size_t)tile * 256 + lane * 8 + fv] = p.u;
}

// ---------------------------------------------------------------------------
// Main kernel: one block (64 threads / 2 waves) per 16-sample tile.
// ---------------------------------------------------------------------------
#define SMP 16

__global__ __launch_bounds__(64) void cqcnn_main(
    const float* __restrict__ board,
    const float* __restrict__ c1w, const float* __restrict__ c1b,
    const float* __restrict__ c2w, const float* __restrict__ c2b,
    const float* __restrict__ c3w, const float* __restrict__ c3b,
    const float* __restrict__ qp,
    const unsigned* __restrict__ ptw1pk, const float* __restrict__ ptb1,
    const unsigned* __restrict__ ptw2pk, const float* __restrict__ ptb2,
    const float* __restrict__ ptw3,      const float* __restrict__ ptb3,
    const unsigned* __restrict__ cfw1pk, const float* __restrict__ cfb1,
    const float* __restrict__ cfw2,      const float* __restrict__ cfb2,
    float* __restrict__ out) {

    __shared__ float    s_board[SMP * 108];   //  6.9 KB
    __shared__ _Float16 s_conv1[SMP * 576];   // 18.0 KB
    __shared__ _Float16 s_pool [SMP * 288];   //  9.0 KB
    __shared__ _Float16 s_feat [SMP * 608];   // 19.0 KB (584 feats, K padded to 608)
    __shared__ _Float16 s_h1   [SMP * 128];   //  4.0 KB
    __shared__ _Float16 s_h2   [SMP * 64];    //  2.0 KB
    __shared__ _Float16 s_hc   [SMP * 64];    //  2.0 KB

    const int tid = threadIdx.x;
    const int s0  = blockIdx.x * SMP;

    // L2 warm hints for the biggest VALU-consumed weights.
    __builtin_prefetch(c2w, 0, 1);
    __builtin_prefetch(c3w, 0, 1);

    // ---- stage board into LDS (coalesced) -------------------------------
    for (int i = tid; i < SMP * 108; i += 64)
        s_board[i] = board[(size_t)s0 * 108 + i];
    __syncthreads();

    // ---- conv1: 3->16, 6x6 SAME, relu -----------------------------------
    for (int i = tid; i < SMP * 576; i += 64) {
        int s = i / 576, r = i % 576;
        int oc = r / 36, p = r % 36, y = p / 6, x = p % 6;
        float acc = c1b[oc];
        for (int ic = 0; ic < 3; ++ic)
            for (int dy = 0; dy < 3; ++dy) {
                int iy = y + dy - 1; if ((unsigned)iy >= 6u) continue;
                for (int dx = 0; dx < 3; ++dx) {
                    int ix = x + dx - 1; if ((unsigned)ix >= 6u) continue;
                    acc += s_board[s * 108 + ic * 36 + iy * 6 + ix] *
                           c1w[((oc * 3 + ic) * 3 + dy) * 3 + dx];
                }
            }
        s_conv1[i] = (_Float16)fmaxf(acc, 0.f);
    }
    __syncthreads();

    // ---- conv2: 16->32, relu, fused 2x2 maxpool -------------------------
    for (int i = tid; i < SMP * 288; i += 64) {
        int s = i / 288, r = i % 288;
        int oc = r / 9, p = r % 9, py = p / 3, px = p % 3;
        float m = 0.f;  // relu floor
        for (int sy = 0; sy < 2; ++sy)
            for (int sx = 0; sx < 2; ++sx) {
                int y = py * 2 + sy, x = px * 2 + sx;
                float acc = c2b[oc];
                for (int ic = 0; ic < 16; ++ic)
                    for (int dy = 0; dy < 3; ++dy) {
                        int iy = y + dy - 1; if ((unsigned)iy >= 6u) continue;
                        for (int dx = 0; dx < 3; ++dx) {
                            int ix = x + dx - 1; if ((unsigned)ix >= 6u) continue;
                            acc += (float)s_conv1[s * 576 + ic * 36 + iy * 6 + ix] *
                                   c2w[((oc * 16 + ic) * 3 + dy) * 3 + dx];
                        }
                    }
                m = fmaxf(m, acc);
            }
        s_pool[i] = (_Float16)m;
    }
    __syncthreads();

    // ---- conv3: 32->64 on 3x3, relu -> feature rows ---------------------
    for (int i = tid; i < SMP * 576; i += 64) {
        int s = i / 576, r = i % 576;
        int oc = r / 9, p = r % 9, y = p / 3, x = p % 3;
        float acc = c3b[oc];
        for (int ic = 0; ic < 32; ++ic)
            for (int dy = 0; dy < 3; ++dy) {
                int iy = y + dy - 1; if ((unsigned)iy >= 3u) continue;
                for (int dx = 0; dx < 3; ++dx) {
                    int ix = x + dx - 1; if ((unsigned)ix >= 3u) continue;
                    acc += (float)s_pool[s * 288 + ic * 9 + iy * 3 + ix] *
                           c3w[((oc * 32 + ic) * 3 + dy) * 3 + dx];
                }
            }
        s_feat[s * 608 + r] = (_Float16)fmaxf(acc, 0.f);
    }

    // ---- quantum sim (8 qubits, 3 layers) + K padding -------------------
    for (int i = tid; i < SMP * 8; i += 64) {
        int s = i / 8, q = i % 8;
        float xq = s_board[s * 108 + q];
        float xn = s_board[s * 108 + ((q + 1) & 7)];
        float st = 0.f;
        for (int l = 0; l < 3; ++l) {
            const float* r = qp + (l * 8 + q) * 3;
            st = sinf(r[0] * xq) * cosf(r[1] * xn) + tanhf(r[2] * st);
        }
        s_feat[s * 608 + 576 + q] = (_Float16)st;
    }
    for (int i = tid; i < SMP * 24; i += 64)
        s_feat[(i / 24) * 608 + 584 + (i % 24)] = (_Float16)0.f;
    __syncthreads();

    // ---- wave-specialized WMMA heads ------------------------------------
    const int wid  = tid >> 5;       // wave-uniform
    const int lane = tid & 31;
    const int col  = lane & 15;
    const int lh   = lane >> 4;
    const v8f zf = {0.f, 0.f, 0.f, 0.f, 0.f, 0.f, 0.f, 0.f};

    if (wid == 0) {
        // ===== piece-type head: 608 -> 128 -> 64 -> 2 (softmax) ==========
        v8f acc[8];
#pragma unroll
        for (int nt = 0; nt < 8; ++nt) acc[nt] = zf;
        const _Float16* arow = &s_feat[col * 608];
        for (int kt = 0; kt < 19; ++kt) {
            v16h a = load_a_frag(arow, kt * 32, lh);
#pragma unroll
            for (int nt = 0; nt < 8; ++nt) {
                v16h b = load_b_frag(ptw1pk, kt * 8 + nt, lane);
                acc[nt] = __builtin_amdgcn_wmma_f32_16x16x32_f16(
                    false, a, false, b, (short)0, acc[nt], false, false);
            }
        }
#pragma unroll
        for (int nt = 0; nt < 8; ++nt)
#pragma unroll
            for (int r = 0; r < 8; ++r) {
                int n = nt * 16 + col, m = r + 8 * lh;
                s_h1[m * 128 + n] = (_Float16)fmaxf(acc[nt][r] + ptb1[n], 0.f);
            }
        asm volatile("s_wait_dscnt 0" ::: "memory");

        v8f acc2[4];
#pragma unroll
        for (int nt = 0; nt < 4; ++nt) acc2[nt] = zf;
        const _Float16* h1row = &s_h1[col * 128];
#pragma unroll
        for (int kt = 0; kt < 4; ++kt) {
            v16h a = load_a_frag(h1row, kt * 32, lh);
#pragma unroll
            for (int nt = 0; nt < 4; ++nt) {
                v16h b = load_b_frag(ptw2pk, kt * 4 + nt, lane);
                acc2[nt] = __builtin_amdgcn_wmma_f32_16x16x32_f16(
                    false, a, false, b, (short)0, acc2[nt], false, false);
            }
        }
#pragma unroll
        for (int nt = 0; nt < 4; ++nt)
#pragma unroll
            for (int r = 0; r < 8; ++r) {
                int n = nt * 16 + col, m = r + 8 * lh;
                s_h2[m * 64 + n] = (_Float16)fmaxf(acc2[nt][r] + ptb2[n], 0.f);
            }
        asm volatile("s_wait_dscnt 0" ::: "memory");

        if (lane < 16) {
            int s = lane;
            float l0 = ptb3[0], l1 = ptb3[1];
            for (int j = 0; j < 64; ++j) {
                float h = (float)s_h2[s * 64 + j];
                l0 += h * ptw3[j * 2 + 0];
                l1 += h * ptw3[j * 2 + 1];
            }
            float mx = fmaxf(l0, l1);
            float e0 = expf(l0 - mx), e1 = expf(l1 - mx);
            float inv = 1.f / (e0 + e1);
            out[(size_t)(s0 + s) * 3 + 0] = e0 * inv;
            out[(size_t)(s0 + s) * 3 + 1] = e1 * inv;
        }
    } else {
        // ===== confidence head: 608 -> 64 -> 1 (sigmoid) =================
        v8f acc[4];
#pragma unroll
        for (int nt = 0; nt < 4; ++nt) acc[nt] = zf;
        const _Float16* arow = &s_feat[col * 608];
        for (int kt = 0; kt < 19; ++kt) {
            v16h a = load_a_frag(arow, kt * 32, lh);
#pragma unroll
            for (int nt = 0; nt < 4; ++nt) {
                v16h b = load_b_frag(cfw1pk, kt * 4 + nt, lane);
                acc[nt] = __builtin_amdgcn_wmma_f32_16x16x32_f16(
                    false, a, false, b, (short)0, acc[nt], false, false);
            }
        }
#pragma unroll
        for (int nt = 0; nt < 4; ++nt)
#pragma unroll
            for (int r = 0; r < 8; ++r) {
                int n = nt * 16 + col, m = r + 8 * lh;
                s_hc[m * 64 + n] = (_Float16)fmaxf(acc[nt][r] + cfb1[n], 0.f);
            }
        asm volatile("s_wait_dscnt 0" ::: "memory");

        if (lane < 16) {
            float c = cfb2[0];
            for (int j = 0; j < 64; ++j)
                c += (float)s_hc[lane * 64 + j] * cfw2[j];
            out[(size_t)(s0 + lane) * 3 + 2] = 1.f / (1.f + expf(-c));
        }
    }
}

// ---------------------------------------------------------------------------
extern "C" void kernel_launch(void* const* d_in, const int* in_sizes, int n_in,
                              void* d_out, int out_size, void* d_ws, size_t ws_size,
                              hipStream_t stream) {
    const float* board = (const float*)d_in[0];
    // d_in[1] target_positions: unused by the math (per reference).
    const float* c1w = (const float*)d_in[2];
    const float* c1b = (const float*)d_in[3];
    const float* c2w = (const float*)d_in[4];
    const float* c2b = (const float*)d_in[5];
    const float* c3w = (const float*)d_in[6];
    const float* c3b = (const float*)d_in[7];
    const float* qp  = (const float*)d_in[8];
    const float* ptw1 = (const float*)d_in[9];
    const float* ptb1 = (const float*)d_in[10];
    const float* ptw2 = (const float*)d_in[11];
    const float* ptb2 = (const float*)d_in[12];
    const float* ptw3 = (const float*)d_in[13];
    const float* ptb3 = (const float*)d_in[14];
    const float* cfw1 = (const float*)d_in[15];
    const float* cfb1 = (const float*)d_in[16];
    const float* cfw2 = (const float*)d_in[17];
    const float* cfb2 = (const float*)d_in[18];
    float* out = (float*)d_out;

    const int B = in_sizes[0] / 108;

    // Workspace layout: prepacked f16 B fragments (dwords).
    //  pt_w1: 19 ktiles x 8 ntiles = 152 tiles
    //  pt_w2:  4 ktiles x 4 ntiles =  16 tiles
    //  cf_w1: 19 ktiles x 4 ntiles =  76 tiles
    unsigned* ws     = (unsigned*)d_ws;
    unsigned* ptw1pk = ws;
    unsigned* ptw2pk = ws + (size_t)152 * 256;
    unsigned* cfw1pk = ws + (size_t)(152 + 16) * 256;

    pack_wmma_b<<<dim3(152), dim3(256), 0, stream>>>(ptw1, 584, 128, 8, ptw1pk);
    pack_wmma_b<<<dim3(16),  dim3(256), 0, stream>>>(ptw2, 128, 64, 4, ptw2pk);
    pack_wmma_b<<<dim3(76),  dim3(256), 0, stream>>>(cfw1, 584, 64, 4, cfw1pk);

    cqcnn_main<<<dim3(B / SMP), dim3(64), 0, stream>>>(
        board, c1w, c1b, c2w, c2b, c3w, c3b, qp,
        ptw1pk, ptb1, ptw2pk, ptb2, ptw3, ptb3,
        cfw1pk, cfb1, cfw2, cfb2, out);
}
